// NNConvNodeLayer_46110768890038
// MI455X (gfx1250) — matrix operand order repaired and checked by reading, not compiled, not run on hardware
//
#include <hip/hip_runtime.h>

typedef __attribute__((ext_vector_type(16))) _Float16 v16h;
typedef __attribute__((ext_vector_type(8)))  _Float16 v8h;
typedef __attribute__((ext_vector_type(8)))  float    v8f;

#define N_NODES 50000
#define N_EDGES 800000
#define NODE_IN 16
#define EDGE_IN 19
#define OUT_CH  16
#define WIDTH   64

// packed-weight fragment bases (fragment = 32 lanes x 16 halves = 512 halves)
#define FRAG_W0  0    // 4 frags  (K=32 padded from 19, 4 N-tiles)
#define FRAG_W1  4    // 8 frags  (4 N-tiles x 2 K-steps)
#define FRAG_W2  12   // 8 frags
#define FRAG_LIN 20   // 32 frags (16 N-tiles x 2 K-steps)
#define N_FRAGS  52

static __device__ __forceinline__ v8f wmma_f16(v16h a, v16h b, v8f c) {
    return __builtin_amdgcn_wmma_f32_16x16x32_f16(false, a, false, b, (short)0, c,
                                                  false, false);
}
// zero accumulator -> lowers to inline-0 SRC2 (no C broadcast movs)
static __device__ __forceinline__ v8f wmma_f16_z(v16h a, v16h b) {
    v8f z = {};
    return __builtin_amdgcn_wmma_f32_16x16x32_f16(false, a, false, b, (short)0, z,
                                                  false, false);
}

// order-independent float atomic max via sign-split int atomics (init must be -inf)
static __device__ __forceinline__ void atomicMaxF(float* a, float v) {
    if (v >= 0.0f) atomicMax((int*)a, __float_as_int(v));
    else           atomicMin((unsigned int*)a, __float_as_uint(v));
}

// ---------------------------------------------------------------------------
// Pre-pack the four weight matrices (f32 row-major KxN) into per-lane WMMA
// B-fragment layout, converted to f16:
//   lanes 0-15 hold K=k0..k0+7 and k0+16..k0+23, lanes 16-31 hold K=k0+8..k0+15 and k0+24..k0+31,
//   half j -> K=8h+j (j<8) or 16+8h+(j-8) (j>=8), h=lane/16, column n = tile*16 + lane%16.
// ---------------------------------------------------------------------------
__global__ void prepack_weights(const float* __restrict__ w0,
                                const float* __restrict__ w1,
                                const float* __restrict__ w2,
                                const float* __restrict__ lin_w,
                                _Float16* __restrict__ packed) {
    int gid = blockIdx.x * blockDim.x + threadIdx.x;
    const int TOTAL = N_FRAGS * 32 * 16;
    if (gid >= TOTAL) return;
    int j    = gid & 15;
    int lane = (gid >> 4) & 31;
    int frag = gid >> 9;

    int col16  = lane & 15;
    int klocal = ((lane < 16) ? 0 : 8) + ((j < 8) ? j : j + 8);   // 0..31 within K-block

    const float* W; int N, Kact, tile, kstep;
    if (frag < FRAG_W1)       { W = w0;    N = 64;  Kact = EDGE_IN; tile = frag;               kstep = 0; }
    else if (frag < FRAG_W2)  { W = w1;    N = 64;  Kact = 64;      tile = (frag - FRAG_W1) >> 1; kstep = (frag - FRAG_W1) & 1; }
    else if (frag < FRAG_LIN) { W = w2;    N = 64;  Kact = 64;      tile = (frag - FRAG_W2) >> 1; kstep = (frag - FRAG_W2) & 1; }
    else                      { W = lin_w; N = 256; Kact = 64;      tile = (frag - FRAG_LIN) >> 1; kstep = (frag - FRAG_LIN) & 1; }

    int k = kstep * 32 + klocal;
    int n = tile * 16 + col16;
    float v = (k < Kact) ? W[k * N + n] : 0.0f;
    packed[gid] = (_Float16)v;
}

__global__ void init_agg(float* __restrict__ agg) {
    int gid = blockIdx.x * blockDim.x + threadIdx.x;
    if (gid < N_NODES * OUT_CH) agg[gid] = -__builtin_huge_valf();
}

// ---------------------------------------------------------------------------
// Fused edge pipeline: 1 wave = 16 edges through the full MLP + message +
// scatter-max.  4 waves / workgroup.  Wedge never touches memory: the N-tile
// index of the last GEMM equals the input-row index of the per-edge 16x16
// matrix, so each WMMA D-fragment is consumed in registers.
// K-lo / K-hi tiles are grouped so consecutive WMMAs are independent (no
// accumulate-chain RAW stalls) and share the same A operand.
// ---------------------------------------------------------------------------
__launch_bounds__(128)
__global__ void nnconv_edge_kernel(const float* __restrict__ node_feats,
                                   const int*   __restrict__ edge_index,
                                   const float* __restrict__ edge_feats,
                                   const float* __restrict__ b0,
                                   const float* __restrict__ b1,
                                   const float* __restrict__ b2,
                                   const float* __restrict__ lin_b,
                                   const _Float16* __restrict__ packed,
                                   float* __restrict__ agg) {
    __shared__ _Float16 sH[4][16 * 64];   // per-wave activation tile (f16)
    __shared__ float    sX[4][16 * 16];   // per-wave gathered src features (f32)

    const int lane = threadIdx.x & 31;
    const int wave = threadIdx.x >> 5;
    const int e0   = (blockIdx.x * 4 + wave) * 16;   // 800000/16 tiles exactly

    const int col = lane & 15;     // N column within tile / C-D column
    const int hs  = lane >> 4;     // lane-half select
    const int ka0 = hs ? 8 : 0;    // A-frag K base for this half
    const int m   = col;           // A-matrix row (edge within tile)

    _Float16* H = sH[wave];
    float*    X = sX[wave];

    // gather source-node features (node table is L2-resident: 3.2 MB)
    #pragma unroll
    for (int t = 0; t < 8; ++t) {
        int idx = t * 32 + lane;                  // 0..255
        int el = idx >> 4, c = idx & 15;
        int s = edge_index[e0 + el];              // src row of edge_index
        X[idx] = node_feats[s * NODE_IN + c];
    }

    // ---- A0 fragment straight from edge_feats (K padded 19 -> 32) ----
    v16h a0;
    const float* ef = edge_feats + (size_t)(e0 + m) * EDGE_IN;
    #pragma unroll
    for (int j = 0; j < 16; ++j) {
        int k = (j < 8) ? (ka0 + j) : (ka0 + 8 + j);
        a0[j] = (_Float16)((k < EDGE_IN) ? ef[k] : 0.0f);
    }

    // ---- stage 1: relu(edge_feats @ W0 + b0) -> H ----
    {
        v8f acc[4];
        #pragma unroll
        for (int tile = 0; tile < 4; ++tile) {
            v16h b = *(const v16h*)(packed + ((FRAG_W0 + tile) * 32 + lane) * 16);
            acc[tile] = wmma_f16_z(a0, b);
        }
        #pragma unroll
        for (int tile = 0; tile < 4; ++tile) {
            float bv = b0[tile * 16 + col];
            #pragma unroll
            for (int r = 0; r < 8; ++r)
                H[(r + hs * 8) * 64 + tile * 16 + col] =
                    (_Float16)fmaxf(acc[tile][r] + bv, 0.0f);
        }
    }
    __builtin_amdgcn_wave_barrier();

    // ---- stages 2 & 3: relu(H @ W{1,2} + b) -> H (in place; all A loads
    //      precede all stores, and same-wave DS ops are in order) ----
    #pragma unroll
    for (int stage = 0; stage < 2; ++stage) {
        const int    fragBase = stage ? FRAG_W2 : FRAG_W1;
        const float* bias     = stage ? b2 : b1;

        const _Float16* hrow = H + m * 64;
        v8h p0 = *(const v8h*)(hrow + ka0);            // K = ka0..ka0+7
        v8h p1 = *(const v8h*)(hrow + ka0 + 16);       // K = ka0+16..ka0+23
        v8h p2 = *(const v8h*)(hrow + 32 + ka0);
        v8h p3 = *(const v8h*)(hrow + 32 + ka0 + 16);
        v16h alo = __builtin_shufflevector(p0, p1, 0,1,2,3,4,5,6,7,8,9,10,11,12,13,14,15);
        v16h ahi = __builtin_shufflevector(p2, p3, 0,1,2,3,4,5,6,7,8,9,10,11,12,13,14,15);
        __builtin_amdgcn_wave_barrier();

        v8f acc[4];
        #pragma unroll
        for (int tile = 0; tile < 4; ++tile) {         // K-lo group: independent
            v16h blo = *(const v16h*)(packed + ((fragBase + tile * 2 + 0) * 32 + lane) * 16);
            acc[tile] = wmma_f16_z(alo, blo);
        }
        #pragma unroll
        for (int tile = 0; tile < 4; ++tile) {         // K-hi group: accumulate
            v16h bhi = *(const v16h*)(packed + ((fragBase + tile * 2 + 1) * 32 + lane) * 16);
            acc[tile] = wmma_f16(ahi, bhi, acc[tile]);
        }
        #pragma unroll
        for (int tile = 0; tile < 4; ++tile) {
            float bv = bias[tile * 16 + col];
            #pragma unroll
            for (int r = 0; r < 8; ++r)
                H[(r + hs * 8) * 64 + tile * 16 + col] =
                    (_Float16)fmaxf(acc[tile][r] + bv, 0.0f);
        }
        __builtin_amdgcn_wave_barrier();
    }

    // ---- stage 4: Wedge = H @ lin_w + lin_b, consumed in registers:
    //      msg[e][o] += x[e][i] * (Dtile_i[e][o] + lin_b[i*16+o]) ----
    {
        const _Float16* hrow = H + m * 64;
        v8h p0 = *(const v8h*)(hrow + ka0);
        v8h p1 = *(const v8h*)(hrow + ka0 + 16);
        v8h p2 = *(const v8h*)(hrow + 32 + ka0);
        v8h p3 = *(const v8h*)(hrow + 32 + ka0 + 16);
        v16h alo = __builtin_shufflevector(p0, p1, 0,1,2,3,4,5,6,7,8,9,10,11,12,13,14,15);
        v16h ahi = __builtin_shufflevector(p2, p3, 0,1,2,3,4,5,6,7,8,9,10,11,12,13,14,15);

        float msg[8];
        #pragma unroll
        for (int r = 0; r < 8; ++r) msg[r] = 0.0f;
        const float* xrow = X + hs * 8 * 16;   // x[e = r + hs*8][i]

        #pragma unroll
        for (int g = 0; g < 4; ++g) {          // 4 groups of 4 N-tiles
            v8f acc[4];
            #pragma unroll
            for (int t = 0; t < 4; ++t) {      // K-lo: independent WMMAs
                int tile = g * 4 + t;
                v16h blo = *(const v16h*)(packed + ((FRAG_LIN + tile * 2 + 0) * 32 + lane) * 16);
                acc[t] = wmma_f16_z(alo, blo);
            }
            #pragma unroll
            for (int t = 0; t < 4; ++t) {      // K-hi: accumulate
                int tile = g * 4 + t;
                v16h bhi = *(const v16h*)(packed + ((FRAG_LIN + tile * 2 + 1) * 32 + lane) * 16);
                acc[t] = wmma_f16(ahi, bhi, acc[t]);
            }
            #pragma unroll
            for (int t = 0; t < 4; ++t) {
                int tile = g * 4 + t;
                float bv = lin_b[tile * 16 + col];
                #pragma unroll
                for (int r = 0; r < 8; ++r)
                    msg[r] = fmaf(xrow[r * 16 + tile], acc[t][r] + bv, msg[r]);
            }
        }

        // scatter-max into agg (dst row of edge_index)
        #pragma unroll
        for (int r = 0; r < 8; ++r) {
            int e  = e0 + r + hs * 8;
            int dn = edge_index[N_EDGES + e];
            atomicMaxF(agg + dn * OUT_CH + col, msg[r]);
        }
    }
}

// ---------------------------------------------------------------------------
// out = node_feats @ root_w + where(agg==-inf, 0, agg) + bias   (f32, tiny)
// ---------------------------------------------------------------------------
__global__ void nnconv_finalize(const float* __restrict__ node_feats,
                                const float* __restrict__ root_w,
                                const float* __restrict__ bias,
                                const float* __restrict__ agg,
                                float* __restrict__ out) {
    int gid = blockIdx.x * blockDim.x + threadIdx.x;
    if (gid >= N_NODES * OUT_CH) return;
    int n = gid >> 4, o = gid & 15;
    float a = agg[gid];
    if (a == -__builtin_huge_valf()) a = 0.0f;
    float s = bias[o] + a;
    const float* x = node_feats + n * NODE_IN;
    #pragma unroll
    for (int i = 0; i < NODE_IN; ++i) s = fmaf(x[i], root_w[i * OUT_CH + o], s);
    out[gid] = s;
}

extern "C" void kernel_launch(void* const* d_in, const int* in_sizes, int n_in,
                              void* d_out, int out_size, void* d_ws, size_t ws_size,
                              hipStream_t stream) {
    const float* node_feats = (const float*)d_in[0];
    const int*   edge_index = (const int*)  d_in[1];
    const float* edge_feats = (const float*)d_in[2];
    const float* w0    = (const float*)d_in[3];
    const float* b0    = (const float*)d_in[4];
    const float* w1    = (const float*)d_in[5];
    const float* b1    = (const float*)d_in[6];
    const float* w2    = (const float*)d_in[7];
    const float* b2    = (const float*)d_in[8];
    const float* lin_w = (const float*)d_in[9];
    const float* lin_b = (const float*)d_in[10];
    const float* root_w= (const float*)d_in[11];
    const float* bias  = (const float*)d_in[12];
    float* out = (float*)d_out;

    float*    agg    = (float*)d_ws;                                        // 3.2 MB
    _Float16* packed = (_Float16*)((char*)d_ws + (size_t)N_NODES * OUT_CH * sizeof(float));

    init_agg<<<(N_NODES * OUT_CH + 255) / 256, 256, 0, stream>>>(agg);
    prepack_weights<<<(N_FRAGS * 512 + 255) / 256, 256, 0, stream>>>(w0, w1, w2, lin_w, packed);
    nnconv_edge_kernel<<<N_EDGES / 64, 128, 0, stream>>>(
        node_feats, edge_index, edge_feats, b0, b1, b2, lin_b, packed, agg);
    nnconv_finalize<<<(N_NODES * OUT_CH + 255) / 256, 256, 0, stream>>>(
        node_feats, root_w, bias, agg, out);
}
